// LocalFrameAttentionWithDiffuser_46806553592107
// MI455X (gfx1250) — compile-verified, matrix operand order checked by reading, not run on hardware
//
#include <hip/hip_runtime.h>

// ---------------------------------------------------------------------------
// CDNA5 (gfx1250) chunked frame-local attention.
// Precision: bf16 inputs to v_wmma_f32_16x16x32_bf16, fp32 accumulate.
// All intermediates (x, W^T, q, k, v, ctx) live as bf16 in d_ws (48 MB).
// GEMMs: 64x64 tile per wave (16 WMMA accumulators, 0.5 KB loaded per WMMA).
// Attention: K tiles staged LDS via the Tensor Data Mover (tensor_load_to_lds)
// when the builtin is available; manual b128 staging otherwise.
// ---------------------------------------------------------------------------

typedef __attribute__((ext_vector_type(16))) __bf16 v16bf;
typedef __attribute__((ext_vector_type(8)))  float  v8f;
typedef __attribute__((ext_vector_type(4)))  unsigned int u32x4;
typedef __attribute__((ext_vector_type(8)))  int i32x8;
typedef __attribute__((ext_vector_type(4)))  int i32x4;

#define WMMA_BF16(a, b, c) \
  __builtin_amdgcn_wmma_f32_16x16x32_bf16(false, (a), false, (b), (short)0, (c), false, false)

// Load one 16-bit A/B WMMA fragment for this lane:
// p must point at the lane's first 8-element run; second run is at p+16.
// (ISA 16-bit 16x32 layout: V0-3 hold K=kBase..kBase+7, V4-7 hold K=kBase+16..+23)
__device__ __forceinline__ v16bf load_frag(const __bf16* p) {
  union { v16bf v; uint4 u[2]; } r;
  r.u[0] = *reinterpret_cast<const uint4*>(p);
  r.u[1] = *reinterpret_cast<const uint4*>(p + 16);
  return r.v;
}

// ---------------------------------------------------------------------------
// Prep kernels
// ---------------------------------------------------------------------------
__global__ __launch_bounds__(256) void cvt_f32_bf16(const float* __restrict__ in,
                                                    __bf16* __restrict__ out, int n) {
  int i = (blockIdx.x * 256 + threadIdx.x) * 4;
  if (i + 3 < n) {
    float4 f = *reinterpret_cast<const float4*>(in + i);
    out[i + 0] = (__bf16)f.x;
    out[i + 1] = (__bf16)f.y;
    out[i + 2] = (__bf16)f.z;
    out[i + 3] = (__bf16)f.w;
  }
}

// W [1024 x 1024] row-major (K x N) -> Wt bf16 [N x K] so B-fragments load
// contiguously. Coalesced read, scattered 2B writes (4 MB, L2-resident).
__global__ __launch_bounds__(256) void cvt_w_transpose(const float* __restrict__ w,
                                                       __bf16* __restrict__ wt) {
  int idx = blockIdx.x * 256 + threadIdx.x;   // 0 .. 1M-1
  int k = idx >> 10, n = idx & 1023;
  wt[(size_t)n * 1024 + k] = (__bf16)w[idx];
}

// ---------------------------------------------------------------------------
// GEMM: C[M x N] = A[M x K] * B[K x N], B supplied pre-transposed Bt[N x K].
// One wave per 64x64 output tile: 4x4 WMMA accumulators, 8 fragment loads
// feed 16 WMMAs per k-step.
// ---------------------------------------------------------------------------
__global__ __launch_bounds__(256) void gemm_bf16_bf16(const __bf16* __restrict__ A,
                                                      const __bf16* __restrict__ Bt,
                                                      __bf16* __restrict__ C,
                                                      int M, int N, int K) {
  const int lane = threadIdx.x & 31, wave = threadIdx.x >> 5;
  const int tilesN = N >> 6;
  const int tile = blockIdx.x * 8 + wave;
  const int tm = (tile / tilesN) << 6;
  const int tn = (tile % tilesN) << 6;
  if (tm >= M) return;
  const int r = lane & 15, kOff = (lane >> 4) << 3;

  v8f acc[4][4] = {};
  const __bf16* ap0 = A + (size_t)(tm + r) * K + kOff;
  const __bf16* bp0 = Bt + (size_t)(tn + r) * K + kOff;

  for (int k0 = 0; k0 < K; k0 += 32) {
    v16bf af[4], bf_[4];
#pragma unroll
    for (int i = 0; i < 4; ++i) {
      af[i]  = load_frag(ap0 + (size_t)(16 * i) * K + k0);
      bf_[i] = load_frag(bp0 + (size_t)(16 * i) * K + k0);
    }
#pragma unroll
    for (int i = 0; i < 4; ++i)
#pragma unroll
      for (int j = 0; j < 4; ++j)
        acc[i][j] = WMMA_BF16(af[i], bf_[j], acc[i][j]);
  }

  const int hi = lane >> 4, col = lane & 15;  // C layout: row = v + 8*hi, col = lane&15
#pragma unroll
  for (int i = 0; i < 4; ++i)
#pragma unroll
    for (int j = 0; j < 4; ++j)
#pragma unroll
      for (int v = 0; v < 8; ++v)
        C[(size_t)(tm + 16 * i + v + 8 * hi) * N + tn + 16 * j + col] = (__bf16)acc[i][j][v];
}

// Output projection: fp32 result + bias, written straight to d_out.
__global__ __launch_bounds__(256) void gemm_bf16_f32_bias(const __bf16* __restrict__ A,
                                                          const __bf16* __restrict__ Bt,
                                                          const float* __restrict__ bias,
                                                          float* __restrict__ C,
                                                          int M, int N, int K) {
  const int lane = threadIdx.x & 31, wave = threadIdx.x >> 5;
  const int tilesN = N >> 6;
  const int tile = blockIdx.x * 8 + wave;
  const int tm = (tile / tilesN) << 6;
  const int tn = (tile % tilesN) << 6;
  if (tm >= M) return;
  const int r = lane & 15, kOff = (lane >> 4) << 3;

  v8f acc[4][4] = {};
  const __bf16* ap0 = A + (size_t)(tm + r) * K + kOff;
  const __bf16* bp0 = Bt + (size_t)(tn + r) * K + kOff;

  for (int k0 = 0; k0 < K; k0 += 32) {
    v16bf af[4], bf_[4];
#pragma unroll
    for (int i = 0; i < 4; ++i) {
      af[i]  = load_frag(ap0 + (size_t)(16 * i) * K + k0);
      bf_[i] = load_frag(bp0 + (size_t)(16 * i) * K + k0);
    }
#pragma unroll
    for (int i = 0; i < 4; ++i)
#pragma unroll
      for (int j = 0; j < 4; ++j)
        acc[i][j] = WMMA_BF16(af[i], bf_[j], acc[i][j]);
  }

  const int hi = lane >> 4, col = lane & 15;
  float bcol[4];
#pragma unroll
  for (int j = 0; j < 4; ++j) bcol[j] = bias[tn + 16 * j + col];
#pragma unroll
  for (int i = 0; i < 4; ++i)
#pragma unroll
    for (int j = 0; j < 4; ++j)
#pragma unroll
      for (int v = 0; v < 8; ++v)
        C[(size_t)(tm + 16 * i + v + 8 * hi) * N + tn + 16 * j + col] = acc[i][j][v] + bcol[j];
}

// ---------------------------------------------------------------------------
// Flash-style chunk-local attention.
// Block = 128 threads (4 waves), all on one (chunk c, head h); each wave owns
// 16 query rows. Keys for chunk c are flat tokens [(c-1)*1024, (c+1)*1024)
// (chunk 0: [0, 1024)). 32 keys per iteration, staged through LDS.
// 1/sqrt(HD) = 0.125 is a power of two -> folded losslessly into bf16 Q frags.
// ---------------------------------------------------------------------------
__global__ __launch_bounds__(128) void attn_kernel(const __bf16* __restrict__ q,
                                                   const __bf16* __restrict__ k,
                                                   const __bf16* __restrict__ v,
                                                   __bf16* __restrict__ ctx) {
  constexpr int L = 1024, DM = 1024, HD = 64;
  __shared__ __align__(16) __bf16 sK[32][64];     // [key][hd]
  __shared__ __align__(16) __bf16 sVt[64][32];    // [hd][key] (transposed)
  __shared__ __align__(16) __bf16 sP[4][16][32];  // per-wave P restage

  const int t = threadIdx.x;
  const int lane = t & 31, wave = t >> 5;
  const int qb = blockIdx.x & 15;
  const int h  = (blockIdx.x >> 4) & 15;
  const int c  = blockIdx.x >> 8;

  const int qRow = qb * 64 + wave * 16;           // within chunk
  const int qTok = c * L + qRow;
  const int kvTok0 = (c == 0) ? 0 : (c - 1) * L;
  const int nIter  = (c == 0) ? 32 : 64;          // 32 keys / iter

  const int r = lane & 15, hi = lane >> 4, kOff = hi << 3;

  // Q fragments (hd 0-31 and 32-63), pre-scaled by 1/sqrt(HD) (exact in bf16).
  const __bf16* qp = q + (size_t)(qTok + r) * DM + h * HD + kOff;
  v16bf qf0 = load_frag(qp);
  v16bf qf1 = load_frag(qp + 32);
#pragma unroll
  for (int i = 0; i < 16; ++i) {
    qf0[i] = (__bf16)((float)qf0[i] * 0.125f);
    qf1[i] = (__bf16)((float)qf1[i] * 0.125f);
  }

  v8f O0 = {}, O1 = {}, O2 = {}, O3 = {};
  float rowm[8], rowl[8];
#pragma unroll
  for (int i = 0; i < 8; ++i) { rowm[i] = -1e30f; rowl[i] = 0.0f; }

  const int ldKey = t >> 2;          // 0..31
  const int ldHs  = (t & 3) << 4;    // 0,16,32,48

#if __has_builtin(__builtin_amdgcn_tensor_load_to_lds)
  // Tensor DMA descriptor pieces that are loop-invariant (all uniform).
  const unsigned ldsK =
      (unsigned)__builtin_amdgcn_readfirstlane((int)(unsigned long long)&sK[0][0]);
#endif

  for (int it = 0; it < nIter; ++it) {
    const size_t tokBase = (size_t)(kvTok0 + it * 32 + ldKey) * DM + h * HD + ldHs;

#if __has_builtin(__builtin_amdgcn_tensor_load_to_lds)
    // ---- K tile via Tensor Data Mover: 2D tile 64 elem x 32 rows, 2B elems,
    //      row stride 1024 elems, into contiguous LDS. One wave issues. ----
    if (wave == 0) {
      unsigned long long ga =
          (unsigned long long)(const void*)(k + (size_t)(kvTok0 + it * 32) * DM + h * HD);
      unsigned galo = (unsigned)__builtin_amdgcn_readfirstlane((int)(ga & 0xffffffffull));
      unsigned gahi = (unsigned)__builtin_amdgcn_readfirstlane((int)(ga >> 32));
      // Group0: count=1 | lds_addr | global_addr[56:0] | type=2 (bits 127:126)
      u32x4 g0 = { 1u, ldsK, galo, (gahi & 0x01ffffffu) | 0x80000000u };
      // Group1: data_size=2B (bits17:16), tensor_dim0=1024 (bits79:48),
      // tensor_dim1=4096 (bits111:80), tile_dim0=64 (bits127:112),
      // tile_dim1=32 (bits143:128), tensor_dim0_stride=1024 (bits207:160)
      i32x8 g1 = { 0x00010000, 0x04000000, 0x10000000, (64 << 16), 32, 1024, 0, 0 };
      i32x4 gz = { 0, 0, 0, 0 };
#if defined(__clang_major__) && (__clang_major__ >= 23)
      i32x8 gz8 = { 0, 0, 0, 0, 0, 0, 0, 0 };
      __builtin_amdgcn_tensor_load_to_lds(g0, g1, gz, gz, gz8, 0);
#else
      __builtin_amdgcn_tensor_load_to_lds(g0, g1, gz, gz, 0);
#endif
    }
#else
    // ---- manual K tile staging ----
    {
      uint4 x0 = *reinterpret_cast<const uint4*>(k + tokBase);
      uint4 x1 = *reinterpret_cast<const uint4*>(k + tokBase + 8);
      *reinterpret_cast<uint4*>(&sK[ldKey][ldHs])     = x0;
      *reinterpret_cast<uint4*>(&sK[ldKey][ldHs + 8]) = x1;
    }
#endif

    // ---- V tile staged transposed (TDM cannot transpose) ----
    {
      union { uint4 u[2]; __bf16 e[16]; } vv;
      vv.u[0] = *reinterpret_cast<const uint4*>(v + tokBase);
      vv.u[1] = *reinterpret_cast<const uint4*>(v + tokBase + 8);
#pragma unroll
      for (int i = 0; i < 16; ++i) sVt[ldHs + i][ldKey] = vv.e[i];
    }
    // Prefetch next tile into cache while this tile computes (global_prefetch_b8).
    if (it + 1 < nIter) {
      __builtin_prefetch(k + tokBase + 32 * DM, 0, 1);
      __builtin_prefetch(v + tokBase + 32 * DM, 0, 1);
    }
#if __has_builtin(__builtin_amdgcn_tensor_load_to_lds)
    if (wave == 0) __builtin_amdgcn_s_wait_tensorcnt(0);
#endif
    __syncthreads();

    // ---- S = Q K^T for 16 queries x 32 keys (4 WMMAs) ----
    v16bf kb00 = load_frag(&sK[r][kOff]);
    v16bf kb01 = load_frag(&sK[r][32 + kOff]);
    v16bf kb10 = load_frag(&sK[16 + r][kOff]);
    v16bf kb11 = load_frag(&sK[16 + r][32 + kOff]);
    v8f z = {};
    v8f s0 = WMMA_BF16(qf0, kb00, z);  s0 = WMMA_BF16(qf1, kb01, s0);
    v8f s1 = WMMA_BF16(qf0, kb10, z);  s1 = WMMA_BF16(qf1, kb11, s1);

    // ---- online softmax (rows: per-VGPR; cols: per-lane within 16) ----
    float corr[8];
#pragma unroll
    for (int vv_ = 0; vv_ < 8; ++vv_) {
      float a = s0[vv_], b = s1[vv_];
      float mx = fmaxf(a, b);
#pragma unroll
      for (int m = 8; m >= 1; m >>= 1) mx = fmaxf(mx, __shfl_xor(mx, m, 16));
      float nm = fmaxf(rowm[vv_], mx);
      float p0 = __expf(a - nm), p1 = __expf(b - nm);
      float ps = p0 + p1;
#pragma unroll
      for (int m = 8; m >= 1; m >>= 1) ps += __shfl_xor(ps, m, 16);
      float cr = __expf(rowm[vv_] - nm);
      rowl[vv_] = rowl[vv_] * cr + ps;
      rowm[vv_] = nm;
      corr[vv_] = cr;
      s0[vv_] = p0;
      s1[vv_] = p1;
    }
#pragma unroll
    for (int vv_ = 0; vv_ < 8; ++vv_) {
      O0[vv_] *= corr[vv_]; O1[vv_] *= corr[vv_];
      O2[vv_] *= corr[vv_]; O3[vv_] *= corr[vv_];
    }

    // ---- restage P (C-layout -> A-fragment layout) via wave-private LDS ----
#pragma unroll
    for (int vv_ = 0; vv_ < 8; ++vv_) {
      int rr = vv_ + 8 * hi;
      sP[wave][rr][r]      = (__bf16)s0[vv_];
      sP[wave][rr][16 + r] = (__bf16)s1[vv_];
    }
    asm volatile("" ::: "memory");   // keep DS stores before DS loads (same wave: in-order)
    v16bf pf = load_frag(&sP[wave][r][kOff]);

    // ---- O += P V (4 WMMAs over hd tiles) ----
    v16bf vb0 = load_frag(&sVt[r][kOff]);
    v16bf vb1 = load_frag(&sVt[16 + r][kOff]);
    v16bf vb2 = load_frag(&sVt[32 + r][kOff]);
    v16bf vb3 = load_frag(&sVt[48 + r][kOff]);
    O0 = WMMA_BF16(pf, vb0, O0);
    O1 = WMMA_BF16(pf, vb1, O1);
    O2 = WMMA_BF16(pf, vb2, O2);
    O3 = WMMA_BF16(pf, vb3, O3);

    __syncthreads();
  }

  // ---- normalize and write ctx (bf16) ----
#pragma unroll
  for (int vv_ = 0; vv_ < 8; ++vv_) {
    float inv = 1.0f / rowl[vv_];
    size_t base = (size_t)(qTok + vv_ + 8 * hi) * DM + h * HD + r;
    ctx[base +  0] = (__bf16)(O0[vv_] * inv);
    ctx[base + 16] = (__bf16)(O1[vv_] * inv);
    ctx[base + 32] = (__bf16)(O2[vv_] * inv);
    ctx[base + 48] = (__bf16)(O3[vv_] * inv);
  }
}

// ---------------------------------------------------------------------------
// Launch
// ---------------------------------------------------------------------------
extern "C" void kernel_launch(void* const* d_in, const int* in_sizes, int n_in,
                              void* d_out, int out_size, void* d_ws, size_t ws_size,
                              hipStream_t stream) {
  (void)in_sizes; (void)n_in; (void)out_size; (void)ws_size;
  const float* x  = (const float*)d_in[0];
  const float* Wq = (const float*)d_in[1];
  const float* Wk = (const float*)d_in[2];
  const float* Wv = (const float*)d_in[3];
  const float* Wo = (const float*)d_in[4];
  const float* bo = (const float*)d_in[5];
  float* out = (float*)d_out;

  char* ws = (char*)d_ws;
  const size_t MB = 1024 * 1024;
  __bf16* xb  = (__bf16*)(ws + 0 * MB);   // 4096x1024 bf16 (8 MB)
  __bf16* wqt = (__bf16*)(ws + 8 * MB);   // 1024x1024 bf16 transposed (2 MB each)
  __bf16* wkt = (__bf16*)(ws + 10 * MB);
  __bf16* wvt = (__bf16*)(ws + 12 * MB);
  __bf16* wot = (__bf16*)(ws + 14 * MB);
  __bf16* qb_ = (__bf16*)(ws + 16 * MB);  // 8 MB each
  __bf16* kb_ = (__bf16*)(ws + 24 * MB);
  __bf16* vb_ = (__bf16*)(ws + 32 * MB);
  __bf16* ctx = (__bf16*)(ws + 40 * MB);  // total 48 MB

  const int nTok = 4096, D = 1024;

  cvt_f32_bf16<<<(nTok * D) / 1024, 256, 0, stream>>>(x, xb, nTok * D);
  cvt_w_transpose<<<(D * D) / 256, 256, 0, stream>>>(Wq, wqt);
  cvt_w_transpose<<<(D * D) / 256, 256, 0, stream>>>(Wk, wkt);
  cvt_w_transpose<<<(D * D) / 256, 256, 0, stream>>>(Wv, wvt);
  cvt_w_transpose<<<(D * D) / 256, 256, 0, stream>>>(Wo, wot);

  // 64x64 tiles: (4096/64)*(1024/64) = 1024 waves -> 128 blocks of 8 waves
  gemm_bf16_bf16<<<128, 256, 0, stream>>>(xb, wqt, qb_, nTok, D, D);
  gemm_bf16_bf16<<<128, 256, 0, stream>>>(xb, wkt, kb_, nTok, D, D);
  gemm_bf16_bf16<<<128, 256, 0, stream>>>(xb, wvt, vb_, nTok, D, D);

  // C=4 chunks * H=16 heads * (L/64)=16 query blocks = 1024 blocks
  attn_kernel<<<1024, 128, 0, stream>>>(qb_, kb_, vb_, ctx);

  gemm_bf16_f32_bias<<<128, 256, 0, stream>>>(ctx, wot, bo, out, nTok, D, D);
}
